// DecoderLayer_71494025609892
// MI455X (gfx1250) — matrix operand order repaired and checked by reading, not compiled, not run on hardware
//
#include <hip/hip_runtime.h>

// ---------------------------------------------------------------------------
// Types for CDNA5 WMMA (wave32): bf16 16x16x32, f32 accumulate
// ---------------------------------------------------------------------------
typedef __attribute__((ext_vector_type(16))) __bf16       bf16x16;
typedef __attribute__((ext_vector_type(8)))  float        f32x8;
typedef __attribute__((ext_vector_type(4)))  unsigned int u32x4;

union FragU { bf16x16 v; u32x4 q[2]; };
union VecU  { u32x4 q; unsigned short s[8]; };

__device__ __forceinline__ unsigned short f2bf(float f) {
  unsigned u = __builtin_bit_cast(unsigned, f);
  u += 0x7fffu + ((u >> 16) & 1u);           // round-to-nearest-even
  return (unsigned short)(u >> 16);
}

// LDS byte offset of a generic pointer to __shared__ (ISA: LDS aperture
// addresses truncate to addr[31:0] = LDS offset).
__device__ __forceinline__ unsigned lds_off(const void* p) {
  return (unsigned)(unsigned long long)p;
}

// Async DMA: global (16B) -> LDS, tracked by ASYNCcnt. Per-lane addresses.
__device__ __forceinline__ void async_copy_b128(unsigned ldsAddr, const void* gptr) {
  asm volatile("global_load_async_to_lds_b128 %0, %1, off"
               :: "v"(ldsAddr), "v"((unsigned long long)gptr) : "memory");
}
__device__ __forceinline__ void wait_asynccnt0() {
  asm volatile("s_wait_asynccnt 0x0" ::: "memory");
}

// ---------------------------------------------------------------------------
// cvt: fp32 -> bf16
// ---------------------------------------------------------------------------
__global__ void cvt_f32_bf16(const float* __restrict__ in,
                             unsigned short* __restrict__ out, int n) {
  int i = blockIdx.x * blockDim.x + threadIdx.x;
  if (i < n) out[i] = f2bf(in[i]);
}

// ---------------------------------------------------------------------------
// GEMM: C[M,N] = A[M,K] * W[N,K]^T + bias ; optional ReLU; f32 and/or bf16 out
// A, W bf16 row-major. Tiles: 128(M) x 128(N), K-step 32. 256 threads = 8 waves.
// Wave (wm in 0..1, wn in 0..3) owns a 64x32 sub-tile = 4x2 WMMA accumulators.
// Double-buffered LDS filled by async global->LDS DMA (ASYNCcnt), so tile t+1
// streams in while tile t feeds the WMMA pipe.
// ---------------------------------------------------------------------------
__global__ __launch_bounds__(256) void gemm_bf16_wmma(
    const unsigned short* __restrict__ A,   // [M,K]
    const unsigned short* __restrict__ W,   // [N,K]
    const float* __restrict__ bias,         // [N] or null
    float* __restrict__ outF,               // [M,N] or null
    unsigned short* __restrict__ outB,      // [M,N] or null
    int M, int N, int K, int relu)
{
  __shared__ __align__(16) unsigned short As[2][128][40];   // pitch 80B
  __shared__ __align__(16) unsigned short Ws[2][128][40];

  const int tid  = threadIdx.x;
  const int wave = tid >> 5;
  const int lane = tid & 31;
  const int sel  = lane >> 4;        // which 16-lane half
  const int l16  = lane & 15;
  const int wm   = wave & 1;         // 0..1  (64 rows each)
  const int wn   = wave >> 1;        // 0..3  (32 cols each)
  const int mBase = blockIdx.y * 128;
  const int nBase = blockIdx.x * 128;

  // tile loader mapping: 16B chunks
  const int lr = tid >> 2;           // 0..63
  const int lc = (tid & 3) * 8;      // 0,8,16,24

  // per-thread async copy of one (A,W) tile pair into buffer `buf`
  auto issue_tile = [&](int kb, int buf) {
#pragma unroll
    for (int r = 0; r < 2; ++r) {
      int row = lr + r * 64;
      async_copy_b128(lds_off(&As[buf][row][lc]),
                      A + (size_t)(mBase + row) * K + kb + lc);
      async_copy_b128(lds_off(&Ws[buf][row][lc]),
                      W + (size_t)(nBase + row) * K + kb + lc);
    }
  };

  f32x8 acc[4][2];
#pragma unroll
  for (int i = 0; i < 4; ++i)
#pragma unroll
    for (int j = 0; j < 2; ++j)
#pragma unroll
      for (int e = 0; e < 8; ++e) acc[i][j][e] = 0.f;

  const int nK = K / 32;
  issue_tile(0, 0);
  wait_asynccnt0();
  __syncthreads();

  for (int t = 0; t < nK; ++t) {
    const int buf = t & 1;
    if (t + 1 < nK) issue_tile((t + 1) * 32, buf ^ 1);   // overlap DMA w/ WMMA

    bf16x16 afr[4], bfr[2];
#pragma unroll
    for (int mt = 0; mt < 4; ++mt) {       // A frag: lane l16 = row, K split halves
      int row = wm * 64 + mt * 16 + l16;
      FragU u;
      u.q[0] = *(const u32x4*)&As[buf][row][sel * 8];        // K = sel*8 .. +7
      u.q[1] = *(const u32x4*)&As[buf][row][16 + sel * 8];   // K = 16+sel*8 .. +7
      afr[mt] = u.v;
    }
#pragma unroll
    for (int nt = 0; nt < 2; ++nt) {       // B frag: lane l16 = col, K split halves
      int row = wn * 32 + nt * 16 + l16;
      FragU u;
      u.q[0] = *(const u32x4*)&Ws[buf][row][sel * 8];
      u.q[1] = *(const u32x4*)&Ws[buf][row][16 + sel * 8];
      bfr[nt] = u.v;
    }
#pragma unroll
    for (int mt = 0; mt < 4; ++mt)
#pragma unroll
      for (int nt = 0; nt < 2; ++nt)
        acc[mt][nt] = __builtin_amdgcn_wmma_f32_16x16x32_bf16(
            false, afr[mt], false, bfr[nt], (short)0, acc[mt][nt], false, false);

    wait_asynccnt0();     // my writes to buf^1 have landed
    __syncthreads();      // everyone done reading buf / writing buf^1
  }

  // epilogue: C layout -> lane holds one column, 8 rows (sel*8 + i) per VGPR
#pragma unroll
  for (int nt = 0; nt < 2; ++nt) {
    int n = nBase + wn * 32 + nt * 16 + l16;
    float bs = bias ? bias[n] : 0.f;
#pragma unroll
    for (int mt = 0; mt < 4; ++mt) {
#pragma unroll
      for (int i = 0; i < 8; ++i) {
        int m = mBase + wm * 64 + mt * 16 + sel * 8 + i;
        float v = acc[mt][nt][i] + bs;
        if (relu) v = v > 0.f ? v : 0.f;
        size_t idx = (size_t)m * N + n;
        if (outF) outF[idx] = v;
        if (outB) outB[idx] = f2bf(v);
      }
    }
  }
}

// ---------------------------------------------------------------------------
// Flash attention (no mask): one block per (128 q rows, b, h).
// Q/K/V bf16 at [(s*B+b), h*64+hd]. 8 waves x 16 q-rows each. Online softmax.
// Q and K tiles stream in via async global->LDS; V goes through VGPRs because
// it is transposed on the LDS store.
// ---------------------------------------------------------------------------
__global__ __launch_bounds__(256) void flash_attn_bf16(
    const unsigned short* __restrict__ Qg,
    const unsigned short* __restrict__ Kg,
    const unsigned short* __restrict__ Vg,
    unsigned short* __restrict__ Og,
    int S, int B, int H, int D)
{
  __shared__ __align__(16) unsigned short Qs[128][72];
  __shared__ __align__(16) unsigned short Ks[64][72];      // [key][hd]
  __shared__ __align__(16) unsigned short Vt[64][72];      // [hd][key] (transposed)
  __shared__ __align__(16) unsigned short Ps[8][16][72];   // wave-private P tiles

  const int tid  = threadIdx.x;
  const int wave = tid >> 5;
  const int lane = tid & 31;
  const int sel  = lane >> 4;
  const int l16  = lane & 15;
  const int qBase = blockIdx.x * 128;
  const int b = blockIdx.y / H;
  const int h = blockIdx.y % H;
  const int colOff = h * 64;

  // ---- async-load Q tile (128x64), 16B chunks ----
#pragma unroll
  for (int t = 0; t < 4; ++t) {
    int ch = tid + t * 256;                 // 1024 chunks
    int row = ch >> 3, col = (ch & 7) * 8;
    async_copy_b128(lds_off(&Qs[row][col]),
                    Qg + ((size_t)(qBase + row) * B + b) * D + colOff + col);
  }

  f32x8 oacc[4];
  float mrow[8], lrow[8];
#pragma unroll
  for (int nt = 0; nt < 4; ++nt)
#pragma unroll
    for (int e = 0; e < 8; ++e) oacc[nt][e] = 0.f;
#pragma unroll
  for (int i = 0; i < 8; ++i) { mrow[i] = -1e30f; lrow[i] = 0.f; }

  bf16x16 qa[2];
  const int nKB = S / 64;

  for (int kb = 0; kb < nKB; ++kb) {
    // ---- load K (async DMA) and V (transposed via VGPRs) blocks ----
#pragma unroll
    for (int t = 0; t < 2; ++t) {
      int ch = tid + t * 256;               // 512 chunks
      int row = ch >> 3, col = (ch & 7) * 8;
      int tok = kb * 64 + row;
      async_copy_b128(lds_off(&Ks[row][col]),
                      Kg + ((size_t)tok * B + b) * D + colOff + col);
      VecU vv;
      vv.q = *(const u32x4*)(Vg + ((size_t)tok * B + b) * D + colOff + col);
#pragma unroll
      for (int j = 0; j < 8; ++j) Vt[col + j][row] = vv.s[j];
    }
    wait_asynccnt0();                        // Q (first iter) + K landed
    __syncthreads();

    if (kb == 0) {  // Q A-frags, resident for all key blocks
#pragma unroll
      for (int ks = 0; ks < 2; ++ks) {
        int row = wave * 16 + l16;
        FragU u;
        u.q[0] = *(const u32x4*)&Qs[row][ks * 32 + sel * 8];
        u.q[1] = *(const u32x4*)&Qs[row][ks * 32 + 16 + sel * 8];
        qa[ks] = u.v;
      }
    }

    // ---- scores S = Q * K^T (16x64 per wave) ----
    f32x8 sacc[4];
#pragma unroll
    for (int nt = 0; nt < 4; ++nt)
#pragma unroll
      for (int e = 0; e < 8; ++e) sacc[nt][e] = 0.f;
#pragma unroll
    for (int nt = 0; nt < 4; ++nt)
#pragma unroll
      for (int ks = 0; ks < 2; ++ks) {
        FragU u;
        u.q[0] = *(const u32x4*)&Ks[nt * 16 + l16][ks * 32 + sel * 8];
        u.q[1] = *(const u32x4*)&Ks[nt * 16 + l16][ks * 32 + 16 + sel * 8];
        sacc[nt] = __builtin_amdgcn_wmma_f32_16x16x32_bf16(
            false, qa[ks], false, u.v, (short)0, sacc[nt], false, false);
      }

    // ---- online softmax (rows = sel*8 + i, shared by 16 lanes) ----
    float mx[8];
#pragma unroll
    for (int nt = 0; nt < 4; ++nt)
#pragma unroll
      for (int i = 0; i < 8; ++i) sacc[nt][i] *= 0.125f;   // 1/sqrt(64)
#pragma unroll
    for (int i = 0; i < 8; ++i) {
      mx[i] = fmaxf(fmaxf(sacc[0][i], sacc[1][i]), fmaxf(sacc[2][i], sacc[3][i]));
#pragma unroll
      for (int d = 1; d < 16; d <<= 1) mx[i] = fmaxf(mx[i], __shfl_xor(mx[i], d, 32));
    }
    float alpha[8];
#pragma unroll
    for (int i = 0; i < 8; ++i) {
      float mn = fmaxf(mrow[i], mx[i]);
      alpha[i] = __expf(mrow[i] - mn);
      mrow[i] = mn;
    }
    float rs[8];
#pragma unroll
    for (int i = 0; i < 8; ++i) rs[i] = 0.f;
#pragma unroll
    for (int nt = 0; nt < 4; ++nt)
#pragma unroll
      for (int i = 0; i < 8; ++i) {
        float pv = __expf(sacc[nt][i] - mrow[i]);
        sacc[nt][i] = pv;
        rs[i] += pv;
      }
#pragma unroll
    for (int i = 0; i < 8; ++i) {
#pragma unroll
      for (int d = 1; d < 16; d <<= 1) rs[i] += __shfl_xor(rs[i], d, 32);
      lrow[i] = lrow[i] * alpha[i] + rs[i];
    }
#pragma unroll
    for (int nt = 0; nt < 4; ++nt)
#pragma unroll
      for (int i = 0; i < 8; ++i) oacc[nt][i] *= alpha[i];

    // ---- P: C-layout -> A-frag layout via wave-private LDS ----
#pragma unroll
    for (int nt = 0; nt < 4; ++nt)
#pragma unroll
      for (int i = 0; i < 8; ++i)
        Ps[wave][sel * 8 + i][nt * 16 + l16] = f2bf(sacc[nt][i]);
    asm volatile("s_wait_dscnt 0" ::: "memory");

    bf16x16 pa[2];
#pragma unroll
    for (int ks = 0; ks < 2; ++ks) {
      FragU u;
      u.q[0] = *(const u32x4*)&Ps[wave][l16][ks * 32 + sel * 8];
      u.q[1] = *(const u32x4*)&Ps[wave][l16][ks * 32 + 16 + sel * 8];
      pa[ks] = u.v;
    }

    // ---- O += P * V (V transposed tile gives B-frag layout directly) ----
#pragma unroll
    for (int nt = 0; nt < 4; ++nt)
#pragma unroll
      for (int ks = 0; ks < 2; ++ks) {
        FragU u;
        u.q[0] = *(const u32x4*)&Vt[nt * 16 + l16][ks * 32 + sel * 8];
        u.q[1] = *(const u32x4*)&Vt[nt * 16 + l16][ks * 32 + 16 + sel * 8];
        oacc[nt] = __builtin_amdgcn_wmma_f32_16x16x32_bf16(
            false, pa[ks], false, u.v, (short)0, oacc[nt], false, false);
      }
    __syncthreads();
  }

  // ---- write O = oacc / l ----
#pragma unroll
  for (int nt = 0; nt < 4; ++nt) {
#pragma unroll
    for (int i = 0; i < 8; ++i) {
      int s = qBase + wave * 16 + sel * 8 + i;
      float v = oacc[nt][i] / lrow[i];
      Og[((size_t)s * B + b) * D + colOff + nt * 16 + l16] = f2bf(v);
    }
  }
}

// ---------------------------------------------------------------------------
// LayerNorm( A + Bv ) * g + be  ; one block (256 thr) per row, D = 1024
// ---------------------------------------------------------------------------
__global__ __launch_bounds__(256) void ln_residual(
    const float* __restrict__ A, const float* __restrict__ Bv,
    const float* __restrict__ g, const float* __restrict__ be,
    float* __restrict__ outF, unsigned short* __restrict__ outB, int D)
{
  const int row = blockIdx.x, tid = threadIdx.x;
  const int wave = tid >> 5, lane = tid & 31;
  const float* a  = A  + (size_t)row * D;
  const float* bb = Bv + (size_t)row * D;

  float x[4], s = 0.f, s2 = 0.f;
#pragma unroll
  for (int i = 0; i < 4; ++i) {
    int j = tid + i * 256;
    x[i] = a[j] + bb[j];
    s += x[i]; s2 += x[i] * x[i];
  }
#pragma unroll
  for (int d = 1; d < 32; d <<= 1) { s += __shfl_xor(s, d, 32); s2 += __shfl_xor(s2, d, 32); }

  __shared__ float rs[8], rs2[8];
  if (lane == 0) { rs[wave] = s; rs2[wave] = s2; }
  __syncthreads();
  float ts = 0.f, ts2 = 0.f;
#pragma unroll
  for (int w = 0; w < 8; ++w) { ts += rs[w]; ts2 += rs2[w]; }

  const float mean = ts / (float)D;
  const float var  = ts2 / (float)D - mean * mean;
  const float inv  = rsqrtf(var + 1e-5f);
#pragma unroll
  for (int i = 0; i < 4; ++i) {
    int j = tid + i * 256;
    float y = (x[i] - mean) * inv * g[j] + be[j];
    size_t idx = (size_t)row * D + j;
    if (outF) outF[idx] = y;
    if (outB) outB[idx] = f2bf(y);
  }
}

// ---------------------------------------------------------------------------
// Host orchestration
// ---------------------------------------------------------------------------
static void launch_gemm(const unsigned short* A, const unsigned short* W,
                        const float* bias, float* outF, unsigned short* outB,
                        int M, int N, int K, int relu, hipStream_t s) {
  dim3 grid(N / 128, M / 128);
  gemm_bf16_wmma<<<grid, 256, 0, s>>>(A, W, bias, outF, outB, M, N, K, relu);
}

static void launch_cvt(const float* in, unsigned short* out, int n, hipStream_t s) {
  cvt_f32_bf16<<<(n + 255) / 256, 256, 0, s>>>(in, out, n);
}

extern "C" void kernel_launch(void* const* d_in, const int* in_sizes, int n_in,
                              void* d_out, int out_size, void* d_ws, size_t ws_size,
                              hipStream_t stream) {
  const int S = 1024, B = 4, D = 1024, H = 16, DFF = 4096;
  const int M = S * B;                 // 4096 token rows
  const size_t MD  = (size_t)M * D;    // 4096*1024
  const size_t WDD = (size_t)D * D;    // 1024*1024
  const size_t WFD = (size_t)DFF * D;  // 4096*1024

  const float* input = (const float*)d_in[0];
  const float* enc   = (const float*)d_in[1];
  const float* sa_w[4] = {(const float*)d_in[2], (const float*)d_in[4],
                          (const float*)d_in[6], (const float*)d_in[8]};
  const float* sa_b[4] = {(const float*)d_in[3], (const float*)d_in[5],
                          (const float*)d_in[7], (const float*)d_in[9]};
  const float* ca_w[4] = {(const float*)d_in[10], (const float*)d_in[12],
                          (const float*)d_in[14], (const float*)d_in[16]};
  const float* ca_b[4] = {(const float*)d_in[11], (const float*)d_in[13],
                          (const float*)d_in[15], (const float*)d_in[17]};
  const float* w1 = (const float*)d_in[18]; const float* b1 = (const float*)d_in[19];
  const float* w2 = (const float*)d_in[20]; const float* b2 = (const float*)d_in[21];
  const float* ln_g[3] = {(const float*)d_in[22], (const float*)d_in[24], (const float*)d_in[26]};
  const float* ln_b[3] = {(const float*)d_in[23], (const float*)d_in[25], (const float*)d_in[27]};

  // ---- carve workspace ----
  char* p = (char*)d_ws;
  auto give = [&](size_t bytes) -> void* {
    void* r = p; p += (bytes + 255) & ~(size_t)255; return r;
  };
  unsigned short* sa_wb[4]; unsigned short* ca_wb[4];
  for (int i = 0; i < 4; ++i) sa_wb[i] = (unsigned short*)give(WDD * 2);
  for (int i = 0; i < 4; ++i) ca_wb[i] = (unsigned short*)give(WDD * 2);
  unsigned short* w1b  = (unsigned short*)give(WFD * 2);
  unsigned short* w2b  = (unsigned short*)give(WFD * 2);
  unsigned short* x_bf   = (unsigned short*)give(MD * 2);
  unsigned short* enc_bf = (unsigned short*)give(MD * 2);
  unsigned short* q_bf   = (unsigned short*)give(MD * 2);
  unsigned short* k_bf   = (unsigned short*)give(MD * 2);
  unsigned short* v_bf   = (unsigned short*)give(MD * 2);
  unsigned short* at_bf  = (unsigned short*)give(MD * 2);
  unsigned short* ln_bf  = (unsigned short*)give(MD * 2);
  unsigned short* h1_bf  = (unsigned short*)give((size_t)M * DFF * 2);
  float* proj_f = (float*)give(MD * 4);
  float* res1_f = (float*)give(MD * 4);
  float* res2_f = (float*)give(MD * 4);

  // ---- bf16 conversions ----
  launch_cvt(input, x_bf,  (int)MD, stream);
  launch_cvt(enc,   enc_bf,(int)MD, stream);
  for (int i = 0; i < 4; ++i) launch_cvt(sa_w[i], sa_wb[i], (int)WDD, stream);
  for (int i = 0; i < 4; ++i) launch_cvt(ca_w[i], ca_wb[i], (int)WDD, stream);
  launch_cvt(w1, w1b, (int)WFD, stream);
  launch_cvt(w2, w2b, (int)WFD, stream);

  dim3 fgrid(S / 128, B * H);

  // ---- self-attention ----
  launch_gemm(x_bf, sa_wb[0], sa_b[0], nullptr, q_bf, M, D, D, 0, stream);
  launch_gemm(x_bf, sa_wb[1], sa_b[1], nullptr, k_bf, M, D, D, 0, stream);
  launch_gemm(x_bf, sa_wb[2], sa_b[2], nullptr, v_bf, M, D, D, 0, stream);
  flash_attn_bf16<<<fgrid, 256, 0, stream>>>(q_bf, k_bf, v_bf, at_bf, S, B, H, D);
  launch_gemm(at_bf, sa_wb[3], sa_b[3], proj_f, nullptr, M, D, D, 0, stream);
  ln_residual<<<M, 256, 0, stream>>>(input, proj_f, ln_g[0], ln_b[0], res1_f, ln_bf, D);

  // ---- cross-attention: query = enc, key = out(ln1), value = enc ----
  launch_gemm(enc_bf, ca_wb[0], ca_b[0], nullptr, q_bf, M, D, D, 0, stream);
  launch_gemm(ln_bf,  ca_wb[1], ca_b[1], nullptr, k_bf, M, D, D, 0, stream);
  launch_gemm(enc_bf, ca_wb[2], ca_b[2], nullptr, v_bf, M, D, D, 0, stream);
  flash_attn_bf16<<<fgrid, 256, 0, stream>>>(q_bf, k_bf, v_bf, at_bf, S, B, H, D);
  launch_gemm(at_bf, ca_wb[3], ca_b[3], proj_f, nullptr, M, D, D, 0, stream);
  ln_residual<<<M, 256, 0, stream>>>(res1_f, proj_f, ln_g[1], ln_b[1], res2_f, ln_bf, D);

  // ---- FFN ----
  launch_gemm(ln_bf, w1b, b1, nullptr, h1_bf, M, DFF, D, 1, stream);   // + ReLU
  launch_gemm(h1_bf, w2b, b2, proj_f, nullptr, M, D, DFF, 0, stream);
  ln_residual<<<M, 256, 0, stream>>>(res2_f, proj_f, ln_g[2], ln_b[2],
                                     (float*)d_out, nullptr, D);
  (void)in_sizes; (void)n_in; (void)out_size; (void)ws_size;
}
